// InnerSoftShiftTriple_48859547959304
// MI455X (gfx1250) — matrix operand order repaired and checked by reading, not compiled, not run on hardware
//
#include <hip/hip_runtime.h>
#include <stdint.h>

// InnerSoftShiftTriple for MI455X (gfx1250, wave32, WMMA).
// B=4, c2=32, N=4096. Fused attention; softmax uses the exact Cauchy-Schwarz
// bound M_i = ||former_i|| (latter_n columns are unit-or-less norm), so there
// is NO online max / rescale / cross-lane reduction in the hot loop.
// former is pre-scaled by log2(e) so the probabilities are raw v_exp_f32
// (2^x) with the bias folded into the WMMA C operand: no per-element mul/add.
// Inner loop = 5 x v_wmma_f32_16x16x32_f16 per 32 source columns:
//   2 for S-tiles (C = -log2e*M), 2 for acc += P @ latter^T,
//   1 for the softmax denominator  l += P @ ones.

typedef __attribute__((ext_vector_type(16))) _Float16 v16h;
typedef __attribute__((ext_vector_type(8)))  _Float16 v8h;
typedef __attribute__((ext_vector_type(2)))  __fp16   fp16x2;  // cvt_pkrtz result
typedef __attribute__((ext_vector_type(8)))  float    v8f;
typedef __attribute__((ext_vector_type(4)))  uint32_t v4u;

#define NPOS 4096   // H*W
#define NB   4      // batch
#define C2   32     // half channels
#define LOG2E 1.44269504088896340736f

// K-axis permutation induced by the (col, col+16) pair packing of P in LDS.
// sigma(K) = (r&1)*16 + (r>>1) + off(q), K = q*8 + r, off = {0,4,8,12}.
// inv_sigma maps a logical column jloc to the K slot that reads it.
__device__ __forceinline__ int inv_sigma(int jloc) {
    int t  = jloc & 15;
    int hi = jloc >> 4;
    int r  = ((t & 3) << 1) | hi;
    int q  = (((t >> 3) & 1) << 1) | ((t >> 2) & 1);
    return (q << 3) | r;
}

// ---------------------------------------------------------------------------
// Prep: per (b, j) column.
//  - FT   [b][i][c]        f16 : former * log2(e)          (WMMA A operand)
//  - LnT  [b][j][c]        f16 : latter / max(||.||,1e-6)  (WMMA B op, GEMM1)
//  - L16p [b][c][jperm]    f16 : latter, j pre-permuted    (WMMA B op, GEMM2)
//  - flagw[N/32]           u32 : mask bit per position
//  - out channels 0..63   : passthrough former|latter (f32, unscaled)
// ---------------------------------------------------------------------------
__global__ __launch_bounds__(256) void prep_kernel(
    const float* __restrict__ x,        // [B][64][N]
    const int* __restrict__ mask,       // [N]
    _Float16* __restrict__ FT,
    _Float16* __restrict__ LnT,
    _Float16* __restrict__ L16p,
    uint32_t* __restrict__ flagw,
    float* __restrict__ out)            // [B][96][N]
{
    int idx = blockIdx.x * blockDim.x + threadIdx.x;   // B*N threads
    if (idx >= NB * NPOS) return;
    int b = idx >> 12;
    int j = idx & (NPOS - 1);
    int jperm = (j & ~31) + inv_sigma(j & 31);

    const float* xb   = x   + (size_t)b * 64 * NPOS;
    float*       outb = out + (size_t)b * 96 * NPOS;

    float lv[C2];
    float ss = 0.f;
#pragma unroll
    for (int c = 0; c < C2; ++c) {
        float fv = xb[(size_t)c * NPOS + j];          // former
        float tv = xb[(size_t)(C2 + c) * NPOS + j];   // latter
        lv[c] = tv;
        ss += tv * tv;
        outb[(size_t)c * NPOS + j]        = fv;       // passthrough (unscaled)
        outb[(size_t)(C2 + c) * NPOS + j] = tv;
        FT[(size_t)idx * C2 + c]          = (_Float16)(fv * LOG2E);
        L16p[((size_t)b * C2 + c) * NPOS + jperm] = (_Float16)tv;
    }
    float inv = 1.f / fmaxf(sqrtf(ss), 1e-6f);
#pragma unroll
    for (int c = 0; c < C2; ++c)
        LnT[(size_t)idx * C2 + c] = (_Float16)(lv[c] * inv);

    if (idx < NPOS / 32) {
        uint32_t w = 0;
#pragma unroll 8
        for (int p = 0; p < 32; ++p)
            w |= (uint32_t)(mask[idx * 32 + p] >= 1) << p;
        flagw[idx] = w;
    }
}

// ---------------------------------------------------------------------------
// Attention: one wave owns a 16-row i-tile. grid = B * N/16 = 1024 waves.
// ---------------------------------------------------------------------------
__global__ __launch_bounds__(32) void attn_kernel(
    const _Float16* __restrict__ FT,     // [B][N][32] (pre-scaled by log2e)
    const _Float16* __restrict__ LnT,    // [B][N][32]
    const _Float16* __restrict__ L16p,   // [B][32][N] (j permuted per 32-block)
    const uint32_t* __restrict__ flagw,  // [N/32]
    float* __restrict__ out)             // [B][96][N]
{
    __shared__ __align__(64) uint32_t plds[16][16];   // P, (col,col+16) pairs

    const int lane = threadIdx.x & 31;
    const int half = lane >> 4;
    const int l16  = lane & 15;
    const int gw   = blockIdx.x;
    const int b    = gw >> 8;
    const int i0   = (gw & 255) << 4;

    // ---- A operand: scaled former rows in WMMA 16-bit A layout ----
    const _Float16* frow = FT + ((size_t)b * NPOS + (i0 + l16)) * C2;
    v8h flo = *(const v8h*)(frow + 16 * half);
    v8h fhi = *(const v8h*)(frow + 16 * half + 8);
    v16h aF = __builtin_shufflevector(flo, fhi,
                                      0,1,2,3,4,5,6,7,8,9,10,11,12,13,14,15);

    // ---- exact logit bound per row: M' = ||log2e * former_row|| ----
    // (aF is already scaled, so this directly yields the base-2 bias.)
    float ssq = 0.f;
#pragma unroll
    for (int e = 0; e < 16; ++e) { float v = (float)aF[e]; ssq += v * v; }
    ssq += __shfl_xor(ssq, 16, 32);        // combine the two K-halves
    float bound = sqrtf(ssq);              // lane holds M' for row l16
    v8f negM;                              // C-tile: row r+8*half gets -M'_row
#pragma unroll
    for (int r = 0; r < 8; ++r)
        negM[r] = -__shfl(bound, r + 8 * half, 32);

    v16h ones;
#pragma unroll
    for (int e = 0; e < 16; ++e) ones[e] = (_Float16)1.f;

    v8f acc0 = {};   // shift channels 0..15
    v8f acc1 = {};   // shift channels 16..31
    v8f lacc = {};   // softmax denominators (row-sum, replicated over cols)

    for (int j0 = 0; j0 < NPOS; j0 += 32) {
        uint32_t fw = flagw[j0 >> 5];      // uniform -> scalar load

        // B operand (latter_n cols), 32B contiguous per lane
        v16h bn0 = *(const v16h*)(LnT + ((size_t)b * NPOS + j0 + l16) * C2 + 16 * half);
        v16h bn1 = *(const v16h*)(LnT + ((size_t)b * NPOS + j0 + 16 + l16) * C2 + 16 * half);

        // S' = log2e * (former . latter_n) - M'  (bias folded into C operand)
        v8f s0 = __builtin_amdgcn_wmma_f32_16x16x32_f16(false, aF, false, bn0,
                                                        (short)0, negM, false, false);
        v8f s1 = __builtin_amdgcn_wmma_f32_16x16x32_f16(false, aF, false, bn1,
                                                        (short)0, negM, false, false);

        // per-lane AND-mask for the packed (col, col+16) pair
        uint32_t mword = (((fw >> l16) & 1u)        ? 0u : 0x0000FFFFu)
                       | (((fw >> (16 + l16)) & 1u) ? 0u : 0xFFFF0000u);

        // P = 2^(S') = exp(S - M) in (0,1]; masked sources zeroed; pack to LDS
#pragma unroll
        for (int r = 0; r < 8; ++r) {
            fp16x2 ph = __builtin_amdgcn_cvt_pkrtz(__builtin_amdgcn_exp2f(s0[r]),
                                                   __builtin_amdgcn_exp2f(s1[r]));
            uint32_t w;
            __builtin_memcpy(&w, &ph, 4);
            plds[r + 8 * half][l16] = w & mword;
        }

        // read P back in A layout (pair packing absorbed by sigma-permuted K)
        v4u plo = *(const v4u*)&plds[l16][8 * half];
        v4u phi = *(const v4u*)&plds[l16][8 * half + 4];
        v16h pA;
        __builtin_memcpy(&pA, &plo, 16);
        __builtin_memcpy((char*)&pA + 16, &phi, 16);

        // B operand (latter, K = permuted local j), 32B contiguous per lane
        v16h bl0 = *(const v16h*)(L16p + ((size_t)b * C2 + l16) * NPOS + j0 + 16 * half);
        v16h bl1 = *(const v16h*)(L16p + ((size_t)b * C2 + 16 + l16) * NPOS + j0 + 16 * half);

        acc0 = __builtin_amdgcn_wmma_f32_16x16x32_f16(false, pA, false, bl0,
                                                      (short)0, acc0, false, false);
        acc1 = __builtin_amdgcn_wmma_f32_16x16x32_f16(false, pA, false, bl1,
                                                      (short)0, acc1, false, false);
        lacc = __builtin_amdgcn_wmma_f32_16x16x32_f16(false, pA, false, ones,
                                                      (short)0, lacc, false, false);
    }

    // ---- epilogue: normalize, scatter into masked destination rows ----
    uint32_t fwi = flagw[i0 >> 5];         // uniform -> scalar load
    float* outb = out + (size_t)b * 96 * NPOS;
#pragma unroll
    for (int r = 0; r < 8; ++r) {
        int i = i0 + r + 8 * half;
        bool fi = (fwi >> (i & 31)) & 1u;
        float invl = 1.f / lacc[r];
        outb[(size_t)(64 + l16) * NPOS + i] = fi ? acc0[r] * invl : 0.f;
        outb[(size_t)(80 + l16) * NPOS + i] = fi ? acc1[r] * invl : 0.f;
    }
}

// ---------------------------------------------------------------------------
extern "C" void kernel_launch(void* const* d_in, const int* in_sizes, int n_in,
                              void* d_out, int out_size, void* d_ws, size_t ws_size,
                              hipStream_t stream) {
    (void)in_sizes; (void)n_in; (void)out_size; (void)ws_size;
    const float* x    = (const float*)d_in[0];   // [4,64,64,64] f32
    const int*   mask = (const int*)d_in[1];     // [64,64] i32
    float*       out  = (float*)d_out;           // [4,96,64,64] f32

    const size_t tensorElems = (size_t)NB * NPOS * C2;     // 512K halves each
    _Float16* FT    = (_Float16*)d_ws;
    _Float16* LnT   = FT  + tensorElems;
    _Float16* L16p  = LnT + tensorElems;
    uint32_t* flagw = (uint32_t*)(L16p + tensorElems);     // 3 MB + 512 B

    prep_kernel<<<(NB * NPOS + 255) / 256, 256, 0, stream>>>(
        x, mask, FT, LnT, L16p, flagw, out);
    attn_kernel<<<NB * (NPOS / 16), 32, 0, stream>>>(
        FT, LnT, L16p, flagw, out);
}
